// TorchGeometricResBlock_54700703481942
// MI455X (gfx1250) — matrix operand order repaired and checked by reading, not compiled, not run on hardware
//
#include <hip/hip_runtime.h>
#include <hip/hip_bf16.h>

typedef __attribute__((ext_vector_type(16))) _Float16 v16h;
typedef __attribute__((ext_vector_type(8)))  _Float16 v8h;
typedef __attribute__((ext_vector_type(8)))  float    v8f;

#define C_DIM 128
#define T_DIM 512
#define FC_DIM 32

// Branch-free GELU (sigmoid form). Lowers to v_exp_f32 + v_rcp_f32 only.
__device__ __forceinline__ float gelu_f(float x) {
    float u = x * (0.7978845608028654f + 0.035677408136300125f * x * x);
    float s = __builtin_amdgcn_rcpf(1.0f + __expf(-2.0f * u));
    return x * s;
}

// ---------------------------------------------------------------------------
// WMMA fragment layouts (wave32, V_WMMA_F32_16X16X32_F16)
// A (16x32 f16): lane l: m=l&15, half=l>>4. a[0..7] = row m, K=half*8..+7;
//                a[8..15] = row m, K=16+half*8..+7  -> two 16B contiguous runs.
// B (32x16 f16): pre-swizzled fragment-major: lane's 16 halves contiguous.
// C/D (16x16 f32): VGPR r: M = r + (l>>4)*8, N = l&15.
// ---------------------------------------------------------------------------
union AFrag { v16h v; v8h h[2]; };

__device__ __forceinline__ v16h load_a_frag(const _Float16* A, int ld) {
    int lane = threadIdx.x & 31;
    int m = lane & 15, half = lane >> 4;
    const _Float16* row = A + m * ld + half * 8;
    AFrag u;
    u.h[0] = *(const v8h*)(row);        // K = half*8 .. +7
    u.h[1] = *(const v8h*)(row + 16);   // K = 16 + half*8 .. +7
    return u.v;
}

__device__ __forceinline__ v16h load_b_frag_packed(const _Float16* P) {
    int lane = threadIdx.x & 31;
    return *(const v16h*)(P + lane * 16);   // 32B contiguous per lane
}

// ---------------------------------------------------------------------------
// Pack weight W[K][Nc] (f32 row-major) into fragment-major f16 tiles:
// out[((kt*(Nc/16)+nt)*32 + lane)*16 + j] = W[kt*32 + (lane>>4)*8 + (j&7) + (j>>3)*16][nt*16 + (lane&15)]
// ---------------------------------------------------------------------------
__global__ void pack_b_kernel(const float* __restrict__ W, _Float16* __restrict__ out,
                              int K, int Nc) {
    int idx = blockIdx.x * 256 + threadIdx.x;
    if (idx >= K * Nc) return;
    int j    = idx & 15;
    int lane = (idx >> 4) & 31;
    int tile = idx >> 9;
    int ntiles = Nc >> 4;
    int kt = tile / ntiles, nt = tile - kt * ntiles;
    int n = lane & 15, kb = (lane >> 4) * 8;
    out[idx] = (_Float16)W[(kt * 32 + kb + (j & 7) + (j >> 3) * 16) * Nc + nt * 16 + n];
}

// ---------------------------------------------------------------------------
// tm[b][j] = sum_k gelu(t[b][k]) * w[k][j] + bias[j]   (B x 256, tiny)
// ---------------------------------------------------------------------------
__global__ void __launch_bounds__(256) time_mod_kernel(
    const float* __restrict__ t, const float* __restrict__ w,
    const float* __restrict__ bias, float* __restrict__ tm, int T) {
    __shared__ float sg[T_DIM];
    int b = blockIdx.x;
    int j = threadIdx.x;
    for (int k = threadIdx.x; k < T; k += 256) sg[k] = gelu_f(t[b * T + k]);
    __syncthreads();
    float acc = bias[j];
    for (int k = 0; k < T; ++k) acc += sg[k] * w[k * 256 + j];
    tm[b * 256 + j] = acc;
}

// ---------------------------------------------------------------------------
// LayerNorm + AdaLN modulation. One wave per row (128 ch = 32 lanes x float4).
// ---------------------------------------------------------------------------
__global__ void __launch_bounds__(256) modulate_kernel(
    const float* __restrict__ f, const float* __restrict__ ln_s,
    const float* __restrict__ ln_b, const float* __restrict__ tm,
    int q, int N, float* __restrict__ out_f32, _Float16* __restrict__ out_f16) {
    int lane = threadIdx.x & 31, wid = threadIdx.x >> 5;
    int node = blockIdx.x * 8 + wid;
    if (node >= N) return;
    int b = node / q;
    const float4 v = ((const float4*)(f + (size_t)node * C_DIM))[lane];
    float s  = v.x + v.y + v.z + v.w;
    float s2 = v.x * v.x + v.y * v.y + v.z * v.z + v.w * v.w;
#pragma unroll
    for (int m = 16; m >= 1; m >>= 1) {
        s  += __shfl_xor(s, m, 32);
        s2 += __shfl_xor(s2, m, 32);
    }
    float mu  = s * (1.0f / C_DIM);
    float var = s2 * (1.0f / C_DIM) - mu * mu;
    float inv = __builtin_amdgcn_rsqf(var + 1e-5f);   // v_rsq_f32
    const float* tmrow = tm + b * 256;
    float r[4] = {v.x, v.y, v.z, v.w};
#pragma unroll
    for (int j = 0; j < 4; ++j) {
        int c = lane * 4 + j;
        float hn  = (r[j] - mu) * inv * ln_s[c] + ln_b[c];
        float val = hn * (1.0f + tmrow[c]) + tmrow[C_DIM + c];
        if (out_f32) out_f32[(size_t)node * C_DIM + c] = val;
        else         out_f16[(size_t)node * C_DIM + c] = (_Float16)val;
    }
}

// ---------------------------------------------------------------------------
// Edge kernel: per wave, 16 edges. 2->32 (VALU), 32->32 (2 WMMA),
// 32->128 (8 WMMA), fused gather * h[src] + atomic scatter to acc[dst].
// Weights arrive pre-swizzled (fragment-major) and are staged in LDS.
// ---------------------------------------------------------------------------
__global__ void __launch_bounds__(256) edge_conv_kernel(
    const float* __restrict__ edge_attr,
    const int* __restrict__ src, const int* __restrict__ dst,
    const float* __restrict__ kw1, const float* __restrict__ kb1,
    const _Float16* __restrict__ w2p, const float* __restrict__ kb2,
    const _Float16* __restrict__ w3p, const float* __restrict__ kb3,
    const float* __restrict__ h1, float* __restrict__ acc, int E) {
    __shared__ _Float16 sW2[2 * 512];        // 2 fragment tiles
    __shared__ _Float16 sW3[8 * 512];        // 8 fragment tiles
    __shared__ _Float16 sK[8][16 * 32];      // per-wave activation staging
    for (int i = threadIdx.x; i < 2 * 512; i += 256) sW2[i] = w2p[i];
    for (int i = threadIdx.x; i < 8 * 512; i += 256) sW3[i] = w3p[i];
    __syncthreads();

    int lane = threadIdx.x & 31, wid = threadIdx.x >> 5;
    int e0 = (blockIdx.x * 8 + wid) * 16;
    if (e0 >= E) return;

    // stage 1: k1 = gelu(attr @ w1 + b1); lane owns feature column `lane`
    float w1x = kw1[lane], w1y = kw1[FC_DIM + lane], bb1 = kb1[lane];
#pragma unroll
    for (int e = 0; e < 16; ++e) {
        float ax = edge_attr[(size_t)(e0 + e) * 2 + 0];
        float ay = edge_attr[(size_t)(e0 + e) * 2 + 1];
        sK[wid][e * 32 + lane] = (_Float16)gelu_f(ax * w1x + ay * w1y + bb1);
    }

    // stage 2: k2 = gelu(k1 @ w2 + b2)  [16,32]@[32,32] -> two 16x16 tiles
    v16h a  = load_a_frag(&sK[wid][0], 32);
    v16h bL = load_b_frag_packed(sW2);
    v16h bR = load_b_frag_packed(sW2 + 512);
    v8f c0 = {}, c1 = {};
    c0 = __builtin_amdgcn_wmma_f32_16x16x32_f16(false, a, false, bL, (short)0, c0, false, false);
    c1 = __builtin_amdgcn_wmma_f32_16x16x32_f16(false, a, false, bR, (short)0, c1, false, false);
    int n = lane & 15, hi = lane >> 4;
#pragma unroll
    for (int r = 0; r < 8; ++r) {
        int m = r + hi * 8;
        sK[wid][m * 32 + n]      = (_Float16)gelu_f(c0[r] + kb2[n]);
        sK[wid][m * 32 + 16 + n] = (_Float16)gelu_f(c1[r] + kb2[16 + n]);
    }

    // stage 3: w = k2 @ w3 + b3 [16,32]@[32,128]; fused message + scatter
    v16h a2 = load_a_frag(&sK[wid][0], 32);
    int se[8], de[8];
#pragma unroll
    for (int r = 0; r < 8; ++r) {       // hoist edge index loads
        int e = e0 + r + hi * 8;
        se[r] = src[e]; de[r] = dst[e];
    }
#pragma unroll
    for (int tile = 0; tile < 8; ++tile) {
        v16h bw = load_b_frag_packed(sW3 + tile * 512);
        v8f c = {};
        c = __builtin_amdgcn_wmma_f32_16x16x32_f16(false, a2, false, bw, (short)0, c, false, false);
        int ch = tile * 16 + n;
        float bias = kb3[ch];
#pragma unroll
        for (int r = 0; r < 8; ++r) {
            float msg = (c[r] + bias) * h1[(size_t)se[r] * C_DIM + ch];
            unsafeAtomicAdd(&acc[(size_t)de[r] * C_DIM + ch], msg);
        }
    }
}

// ---------------------------------------------------------------------------
// x = feats + (acc + conv_bias) * rcp(norm_vec)
// ---------------------------------------------------------------------------
__global__ void residual1_kernel(const float* __restrict__ feats,
                                 const float* __restrict__ acc,
                                 const float* __restrict__ conv_bias,
                                 const float* __restrict__ norm_vec,
                                 float* __restrict__ x, int total) {
    int i = blockIdx.x * 256 + threadIdx.x;
    if (i >= total) return;
    int c = i & (C_DIM - 1), node = i >> 7;
    float rn = __builtin_amdgcn_rcpf(norm_vec[node]);   // v_rcp_f32
    x[i] = feats[i] + (acc[i] + conv_bias[c]) * rn;
}

// ---------------------------------------------------------------------------
// FFN1: Y = gelu(h2 @ W1 + b1)  [N,128]@[128,256]
// Wave owns a 16-row strip: A fragments (K=128 -> 4) live in registers,
// loop over 16 N-tiles -> 64 WMMAs per wave, A loaded once.
// ---------------------------------------------------------------------------
__global__ void __launch_bounds__(256) ffn1_kernel(
    const _Float16* __restrict__ A, const _Float16* __restrict__ Wp,
    const float* __restrict__ bias, _Float16* __restrict__ Y, int N) {
    int lane = threadIdx.x & 31, wid = threadIdx.x >> 5;
    int m0 = (blockIdx.x * 8 + wid) * 16;
    if (m0 >= N) return;
    v16h af[4];
#pragma unroll
    for (int ks = 0; ks < 4; ++ks)
        af[ks] = load_a_frag(A + (size_t)m0 * 128 + ks * 32, 128);
    int n = lane & 15, hi = lane >> 4;
#pragma unroll
    for (int nt = 0; nt < 16; ++nt) {
        v8f c = {};
#pragma unroll
        for (int ks = 0; ks < 4; ++ks) {
            v16h bf = load_b_frag_packed(Wp + (size_t)(ks * 16 + nt) * 512);
            c = __builtin_amdgcn_wmma_f32_16x16x32_f16(false, af[ks], false, bf, (short)0, c, false, false);
        }
        float bj = bias[nt * 16 + n];
#pragma unroll
        for (int r = 0; r < 8; ++r) {
            int m = m0 + r + hi * 8;
            Y[(size_t)m * 256 + nt * 16 + n] = (_Float16)gelu_f(c[r] + bj);
        }
    }
}

// ---------------------------------------------------------------------------
// FFN2: out += Y @ W2 + b2  [N,256]@[256,128]
// Wave owns a 16-row strip: 8 A fragments in registers, 8 N-tiles -> 64 WMMAs.
// ---------------------------------------------------------------------------
__global__ void __launch_bounds__(256) ffn2_kernel(
    const _Float16* __restrict__ A, const _Float16* __restrict__ Wp,
    const float* __restrict__ bias, float* __restrict__ out, int N) {
    int lane = threadIdx.x & 31, wid = threadIdx.x >> 5;
    int m0 = (blockIdx.x * 8 + wid) * 16;
    if (m0 >= N) return;
    v16h af[8];
#pragma unroll
    for (int ks = 0; ks < 8; ++ks)
        af[ks] = load_a_frag(A + (size_t)m0 * 256 + ks * 32, 256);
    int n = lane & 15, hi = lane >> 4;
#pragma unroll
    for (int nt = 0; nt < 8; ++nt) {
        v8f c = {};
#pragma unroll
        for (int ks = 0; ks < 8; ++ks) {
            v16h bf = load_b_frag_packed(Wp + (size_t)(ks * 8 + nt) * 512);
            c = __builtin_amdgcn_wmma_f32_16x16x32_f16(false, af[ks], false, bf, (short)0, c, false, false);
        }
        float bj = bias[nt * 16 + n];
#pragma unroll
        for (int r = 0; r < 8; ++r) {
            int m = m0 + r + hi * 8;
            out[(size_t)m * 128 + nt * 16 + n] += c[r] + bj;
        }
    }
}

// ---------------------------------------------------------------------------
extern "C" void kernel_launch(void* const* d_in, const int* in_sizes, int n_in,
                              void* d_out, int out_size, void* d_ws, size_t ws_size,
                              hipStream_t stream) {
    const float* feats      = (const float*)d_in[0];
    const int*   edge_index = (const int*)  d_in[1];
    const float* edge_attr  = (const float*)d_in[2];
    const float* t          = (const float*)d_in[3];
    const float* norm_vec   = (const float*)d_in[4];
    const float* ln1_s      = (const float*)d_in[5];
    const float* ln1_b      = (const float*)d_in[6];
    const float* ln2_s      = (const float*)d_in[7];
    const float* ln2_b      = (const float*)d_in[8];
    const float* tmlp1_w    = (const float*)d_in[9];
    const float* tmlp1_b    = (const float*)d_in[10];
    const float* tmlp2_w    = (const float*)d_in[11];
    const float* tmlp2_b    = (const float*)d_in[12];
    const float* ker_w1     = (const float*)d_in[13];
    const float* ker_b1     = (const float*)d_in[14];
    const float* ker_w2     = (const float*)d_in[15];
    const float* ker_b2     = (const float*)d_in[16];
    const float* ker_w3     = (const float*)d_in[17];
    const float* ker_b3     = (const float*)d_in[18];
    const float* conv_bias  = (const float*)d_in[19];
    const float* mlp_w1     = (const float*)d_in[20];
    const float* mlp_b1     = (const float*)d_in[21];
    const float* mlp_w2     = (const float*)d_in[22];
    const float* mlp_b2     = (const float*)d_in[23];

    const int N = in_sizes[0] / C_DIM;
    const int E = in_sizes[2] / 2;
    const int B = in_sizes[3] / T_DIM;
    const int q = N / B;

    char* ws = (char*)d_ws;
    size_t off = 0;
    float*    acc  = (float*)(ws + off);    off += (size_t)N * C_DIM * 4;
    float*    h1   = (float*)(ws + off);    off += (size_t)N * C_DIM * 4;
    _Float16* h2h  = (_Float16*)(ws + off); off += (size_t)N * C_DIM * 2;
    _Float16* yh   = (_Float16*)(ws + off); off += (size_t)N * 2 * C_DIM * 2;
    float*    tm1  = (float*)(ws + off);    off += (size_t)B * 256 * 4;
    float*    tm2  = (float*)(ws + off);    off += (size_t)B * 256 * 4;
    _Float16* kw2p = (_Float16*)(ws + off); off += (size_t)FC_DIM * FC_DIM * 2;
    _Float16* kw3p = (_Float16*)(ws + off); off += (size_t)FC_DIM * C_DIM * 2;
    _Float16* mw1p = (_Float16*)(ws + off); off += (size_t)C_DIM * 2 * C_DIM * 2;
    _Float16* mw2p = (_Float16*)(ws + off); off += (size_t)2 * C_DIM * C_DIM * 2;

    hipMemsetAsync(acc, 0, (size_t)N * C_DIM * 4, stream);

    // Pre-swizzle all GEMM weights into fragment-major f16
    pack_b_kernel<<<(FC_DIM * FC_DIM + 255) / 256, 256, 0, stream>>>(ker_w2, kw2p, FC_DIM, FC_DIM);
    pack_b_kernel<<<(FC_DIM * C_DIM + 255) / 256, 256, 0, stream>>>(ker_w3, kw3p, FC_DIM, C_DIM);
    pack_b_kernel<<<(C_DIM * 2 * C_DIM + 255) / 256, 256, 0, stream>>>(mlp_w1, mw1p, C_DIM, 2 * C_DIM);
    pack_b_kernel<<<(2 * C_DIM * C_DIM + 255) / 256, 256, 0, stream>>>(mlp_w2, mw2p, 2 * C_DIM, C_DIM);

    time_mod_kernel<<<B, 256, 0, stream>>>(t, tmlp1_w, tmlp1_b, tm1, T_DIM);
    time_mod_kernel<<<B, 256, 0, stream>>>(t, tmlp2_w, tmlp2_b, tm2, T_DIM);

    // h1 = modulate(feats)  [f32]
    modulate_kernel<<<(N + 7) / 8, 256, 0, stream>>>(feats, ln1_s, ln1_b, tm1, q, N, h1, nullptr);

    // edge conv: fused per-edge MLP + gather + scatter-add
    edge_conv_kernel<<<(E + 127) / 128, 256, 0, stream>>>(
        edge_attr, edge_index, edge_index + E,
        ker_w1, ker_b1, kw2p, ker_b2, kw3p, ker_b3, h1, acc, E);

    // x = feats + (acc + bias)/norm  -> stored in d_out
    float* x = (float*)d_out;
    residual1_kernel<<<(N * C_DIM + 255) / 256, 256, 0, stream>>>(
        feats, acc, conv_bias, norm_vec, x, N * C_DIM);

    // h2 = modulate(x)  [f16]
    modulate_kernel<<<(N + 7) / 8, 256, 0, stream>>>(x, ln2_s, ln2_b, tm2, q, N, nullptr, h2h);

    // FFN
    ffn1_kernel<<<(N / 16 + 7) / 8, 256, 0, stream>>>(h2h, mw1p, mlp_b1, yh, N);
    ffn2_kernel<<<(N / 16 + 7) / 8, 256, 0, stream>>>(yh, mw2p, mlp_b2, x, N);
}
